// AttentionCircuit_41747082117363
// MI455X (gfx1250) — matrix operand order repaired and checked by reading, not compiled, not run on hardware
//
#include <hip/hip_runtime.h>
#include <hip/hip_bf16.h>

// ---------------------------------------------------------------------------
// Types for CDNA5 WMMA (wave32): V_WMMA_F32_16X16X32_BF16
// ---------------------------------------------------------------------------
typedef __attribute__((ext_vector_type(16))) __bf16 v16bf;
typedef __attribute__((ext_vector_type(8)))  __bf16 bf16x8;
typedef __attribute__((ext_vector_type(8)))  float  v8f;
typedef __attribute__((ext_vector_type(4)))  unsigned uint32x4;
typedef __attribute__((ext_vector_type(8)))  int      int32x8;
typedef __attribute__((ext_vector_type(4)))  int      int32x4;

#define DEV __device__ __forceinline__

DEV __bf16 f2bf(float f) {
    unsigned u = __builtin_bit_cast(unsigned, f);
    unsigned r = (u + 0x7FFFu + ((u >> 16) & 1u)) >> 16;   // round-to-nearest-even
    return __builtin_bit_cast(__bf16, (unsigned short)r);
}
DEV float bf2f(__bf16 h) {
    unsigned short us = __builtin_bit_cast(unsigned short, h);
    return __builtin_bit_cast(float, (unsigned)us << 16);
}

union Frag16 { v16bf v; bf16x8 h[2]; };

#define HAVE_TDM (__has_builtin(__builtin_amdgcn_tensor_load_to_lds) && \
                  __has_builtin(__builtin_amdgcn_s_wait_tensorcnt))

#if HAVE_TDM
// ---------------------------------------------------------------------------
// Tensor Data Mover: async 2-D tile load (32 x 128 rows of bf16) into LDS.
// D# per CDNA5 ISA ch.8:
//   group0: count=1 | lds_addr | global_addr | type=2
//   group1: data_size=2B, pad_enable, pad_interval=16 DW (one 64B row),
//           pad_amount=4 DW (16B) -> padded LDS row stride of 40 halves,
//           tensor dims huge (tiles never clip), tile 32 x 128, stride0 = K.
// LDS byte offset of a __shared__ object == low 32 bits of its flat address.
// This toolchain exposes the 6-arg builtin:
//   (uint32x4 g0, int32x8 g1, int32x4 g2, int32x4 g3, int32x8 g4, i32 cpol)
// groups 2..4 are unused for a 2-D tensor -> zero-filled.
// ---------------------------------------------------------------------------
DEV void tdm_load_tile(unsigned lds_off, const __bf16* gptr, int k_stride) {
    unsigned long long ga = (unsigned long long)(uintptr_t)gptr;
    uint32x4 g0;
    g0[0] = 1u;                                    // count=1, load, no gather
    g0[1] = lds_off;                               // LDS address (bytes)
    g0[2] = (unsigned)ga;                          // global_addr[31:0]
    g0[3] = (unsigned)(ga >> 32) | (2u << 30);     // global_addr[56:32] | type=2
    int32x8 g1;
    g1[0] = (int)((1u << 16)      // data_size = 2 bytes
                | (1u << 20)      // pad_enable
                | (3u << 22)      // pad_interval: 16 dwords (64B = one tile row)
                | (3u << 25));    // pad_amount:   4 dwords (16B pad)
    const unsigned td = 1u << 20;                  // huge tensor dims: no clip
    g1[1] = (int)((td & 0xFFFFu) << 16);           // tensor_dim0 lo
    g1[2] = (int)((td >> 16) | ((td & 0xFFFFu) << 16));   // dim0 hi | dim1 lo
    g1[3] = (int)((td >> 16) | (32u << 16));       // dim1 hi | tile_dim0 = 32
    g1[4] = (int)128u;                             // tile_dim1 = 128, tile_dim2=0
    g1[5] = k_stride;                              // tensor_dim0_stride lo32
    g1[6] = 0;                                     // stride0 hi, stride1 lo
    g1[7] = 0;                                     // stride1 hi
    int32x4 gz4 = {0, 0, 0, 0};                    // 2-D: groups 2/3 unused
    int32x8 gz8 = {0, 0, 0, 0, 0, 0, 0, 0};
    __builtin_amdgcn_tensor_load_to_lds(g0, g1, gz4, gz4, gz8, 0);
}
#endif

// ---------------------------------------------------------------------------
// Conversion / packing kernels (f32 -> bf16; every GEMM B operand stored NxK)
// ---------------------------------------------------------------------------
__global__ void cast_f32_to_bf16(const float* __restrict__ in,
                                 __bf16* __restrict__ out, int n) {
    int i = blockIdx.x * 256 + threadIdx.x;
    if (i < n) out[i] = f2bf(in[i]);
}

// f_neurons (64,1024,64) [n][d][r] -> FbT (4096,1024): FbT[c][d], c=n*64+r
__global__ void pack_f_neurons(const float* __restrict__ f,
                               __bf16* __restrict__ FbT) {
    int i = blockIdx.x * 256 + threadIdx.x;          // 4096*1024
    int d = i & 1023, c = i >> 10;
    int n = c >> 6, rr = c & 63;
    FbT[i] = f2bf(f[((size_t)n << 16) + (d << 6) + rr]);
}

// r_neurons (64,64,1024) [n][r][d] -> RT (1024,2048): RT[d][c], c=n*64+r
__global__ void pack_r_neurons(const float* __restrict__ rn,
                               __bf16* __restrict__ RT, int nbase) {
    int i = blockIdx.x * 256 + threadIdx.x;          // 1024*2048
    int cc = i & 2047, d = i >> 11;
    int n = cc >> 6, rr = cc & 63;
    RT[i] = f2bf(rn[((size_t)(nbase + n) << 16) + (rr << 10) + d]);
}

// ---------------------------------------------------------------------------
// bf16 WMMA GEMM:  C(MxN) = A(MxK, row-major) * Bt(NxK)^T
// Block tile 128x128, K-step 32, double-buffered LDS fed by the TDM.
// 8 waves in a 4(M) x 2(N) grid; each wave: 2 A-frags x 4 B-frags
// = 8 v_wmma_f32_16x16x32_bf16 per K-step (B fragment reused across A rows).
// ---------------------------------------------------------------------------
template <bool OUT_BF16>
__global__ __launch_bounds__(256) void gemm_bf16_kernel(
    const __bf16* __restrict__ A, const __bf16* __restrict__ Bt,
    float* __restrict__ Cf, __bf16* __restrict__ Cb, int M, int N, int K) {
    constexpr int BM = 128, BN = 128, BK = 32, LDT = 40;   // padded halves/row
    __shared__ alignas(16) __bf16 As[2][BM * LDT + 8];
    __shared__ alignas(16) __bf16 Bs[2][BN * LDT + 8];

    const int tid = threadIdx.x, lane = tid & 31, w = tid >> 5;
    const int wm = w & 3, wn = w >> 2;
    const int m0 = blockIdx.y * BM, n0 = blockIdx.x * BN;
    const int r = lane & 15, hi = lane >> 4;

    const __bf16* Ag = A  + (size_t)m0 * K;
    const __bf16* Bg = Bt + (size_t)n0 * K;

    v8f c[2][4] = {};
    const int niter = K / BK;

#if HAVE_TDM
    if      (w == 0) tdm_load_tile((unsigned)(uintptr_t)&As[0][0], Ag, K);
    else if (w == 1) tdm_load_tile((unsigned)(uintptr_t)&Bs[0][0], Bg, K);
#else
    {   // manual staging fallback: 16 halves per thread
        const int row = tid >> 1, col = (tid & 1) * 16;
        const bf16x8* ap = (const bf16x8*)(Ag + (size_t)row * K + col);
        const bf16x8* bp = (const bf16x8*)(Bg + (size_t)row * K + col);
        *(bf16x8*)&As[0][row * LDT + col]     = ap[0];
        *(bf16x8*)&As[0][row * LDT + col + 8] = ap[1];
        *(bf16x8*)&Bs[0][row * LDT + col]     = bp[0];
        *(bf16x8*)&Bs[0][row * LDT + col + 8] = bp[1];
    }
#endif

    for (int i = 0; i < niter; ++i) {
        const int cur = i & 1, nxt = cur ^ 1;
        if (i + 1 < niter) {
            const int k1 = (i + 1) * BK;
#if HAVE_TDM
            if      (w == 0) tdm_load_tile((unsigned)(uintptr_t)&As[nxt][0], Ag + k1, K);
            else if (w == 1) tdm_load_tile((unsigned)(uintptr_t)&Bs[nxt][0], Bg + k1, K);
            __builtin_amdgcn_s_wait_tensorcnt((short)1);   // cur tile landed
#else
            const int row = tid >> 1, col = (tid & 1) * 16;
            const bf16x8* ap = (const bf16x8*)(Ag + (size_t)row * K + k1 + col);
            const bf16x8* bp = (const bf16x8*)(Bg + (size_t)row * K + k1 + col);
            *(bf16x8*)&As[nxt][row * LDT + col]     = ap[0];
            *(bf16x8*)&As[nxt][row * LDT + col + 8] = ap[1];
            *(bf16x8*)&Bs[nxt][row * LDT + col]     = bp[0];
            *(bf16x8*)&Bs[nxt][row * LDT + col + 8] = bp[1];
#endif
        }
#if HAVE_TDM
        else {
            __builtin_amdgcn_s_wait_tensorcnt((short)0);   // last tile landed
        }
#endif
        __syncthreads();   // cur buffer ready for every wave

        // A fragments (16x32): lane<16 row=r, k=0..15; lane>=16 k=16..31
        Frag16 a[2];
#pragma unroll
        for (int mi = 0; mi < 2; ++mi) {
            const __bf16* ap = &As[cur][(wm * 32 + mi * 16 + r) * LDT];
            a[mi].h[0] = *(const bf16x8*)(ap + hi * 16);
            a[mi].h[1] = *(const bf16x8*)(ap + hi * 16 + 8);
        }
#pragma unroll
        for (int j = 0; j < 4; ++j) {
            // B fragment (32x16): lane holds column n = j*16+r, 16 contig k
            Frag16 b;
            const __bf16* bp = &Bs[cur][(wn * 64 + j * 16 + r) * LDT + hi * 16];
            b.h[0] = *(const bf16x8*)bp;
            b.h[1] = *(const bf16x8*)(bp + 8);
#pragma unroll
            for (int mi = 0; mi < 2; ++mi)
                c[mi][j] = __builtin_amdgcn_wmma_f32_16x16x32_bf16(
                    false, a[mi].v, false, b.v, (short)0, c[mi][j], false, false);
        }
        __syncthreads();   // all waves done with cur before it is refilled
    }

#pragma unroll
    for (int mi = 0; mi < 2; ++mi)
#pragma unroll
        for (int j = 0; j < 4; ++j) {
            const int col = n0 + wn * 64 + j * 16 + r;
#pragma unroll
            for (int v = 0; v < 8; ++v) {         // C layout: row = v + 8*hi
                const int row = m0 + wm * 32 + mi * 16 + hi * 8 + v;
                if (OUT_BF16) Cb[(size_t)row * N + col] = f2bf(c[mi][j][v]);
                else          Cf[(size_t)row * N + col] = c[mi][j][v];
            }
        }
}

// ---------------------------------------------------------------------------
// Per-token combine:  h = weighted sum over n of Y slices; then build the
// rank-1 token matrices A_Q/A_K/A_V[s, n*64+r] = w[s,n] * h[s,r]  (bf16)
// ---------------------------------------------------------------------------
__global__ __launch_bounds__(256) void combine_kernel(
    const __bf16* __restrict__ Y,
    const float* __restrict__ fqkw, const float* __restrict__ fvw,
    const float* __restrict__ wq,   const float* __restrict__ wk,
    const float* __restrict__ wv,
    __bf16* __restrict__ AQ, __bf16* __restrict__ AK, __bf16* __restrict__ AV) {
    const int s = blockIdx.x, t = threadIdx.x;
    __shared__ float sw[160];
    __shared__ float h[128];
    if (t < 32) {
        sw[t]       = fqkw[s * 32 + t];
        sw[32 + t]  = fvw [s * 32 + t];
        sw[64 + t]  = wq  [s * 32 + t];
        sw[96 + t]  = wk  [s * 32 + t];
        sw[128 + t] = wv  [s * 32 + t];
    }
    __syncthreads();
    if (t < 128) {
        const int rr = t & 63, bank = t >> 6;     // bank 0 = qk, 1 = v
        const __bf16* yrow = Y + (size_t)s * 4096 + bank * 2048 + rr;
        float acc = 0.f;
#pragma unroll
        for (int n = 0; n < 32; ++n) acc += sw[bank * 32 + n] * bf2f(yrow[n * 64]);
        h[t] = acc;
    }
    __syncthreads();
#pragma unroll
    for (int i = 0; i < 8; ++i) {
        const int cidx = t * 8 + i, n = cidx >> 6, rr = cidx & 63;
        const size_t o = (size_t)s * 2048 + cidx;
        AQ[o] = f2bf(sw[64 + n]  * h[rr]);
        AK[o] = f2bf(sw[96 + n]  * h[rr]);
        AV[o] = f2bf(sw[128 + n] * h[64 + rr]);
    }
}

// ---------------------------------------------------------------------------
// Flash attention (causal, dh=64). Block = (b, h, 128 q rows), 8 waves,
// each wave owns 16 q rows. Key blocks of 32: 4 score WMMAs + 4 PV WMMAs.
// K staged key-major; V staged transposed (dh-major) so PV B-fragments are
// contiguous; P converted C-layout -> A-layout via per-wave LDS buffer.
// ---------------------------------------------------------------------------
__global__ __launch_bounds__(256) void attn_kernel(
    const __bf16* __restrict__ Qb, const __bf16* __restrict__ Kb,
    const __bf16* __restrict__ Vb, __bf16* __restrict__ AO) {
    constexpr int DM = 1024;
    const int qc = blockIdx.x, hh = blockIdx.y, b = blockIdx.z;
    const int tid = threadIdx.x, lane = tid & 31, w = tid >> 5;
    const int r = lane & 15, hi = lane >> 4;
    const size_t base = ((size_t)b * 2048) * DM + hh * 64;
    const int q0 = qc * 128 + w * 16;

    __shared__ alignas(16) __bf16 Ks[32 * 72];       // [key][dh], pad 8
    __shared__ alignas(16) __bf16 Vt[64 * 40];       // [dh][key], pad 8
    __shared__ alignas(16) __bf16 Ps[8 * 16 * 40];   // per-wave P, pad 8

    // Q fragments (A layout), two dh-chunks of 32, kept in registers
    Frag16 qf[2];
#pragma unroll
    for (int ch = 0; ch < 2; ++ch) {
        const __bf16* qp = Qb + base + (size_t)(q0 + r) * DM + ch * 32;
        qf[ch].h[0] = *(const bf16x8*)(qp + hi * 16);
        qf[ch].h[1] = *(const bf16x8*)(qp + hi * 16 + 8);
    }
    v8f o[4] = {};
    float mrow[8], lrow[8];
#pragma unroll
    for (int v = 0; v < 8; ++v) { mrow[v] = -1e30f; lrow[v] = 0.f; }

    const int nkb = qc * 4 + 4;                  // key blocks this CTA needs
    for (int kb = 0; kb < nkb; ++kb) {
        const int key0 = kb * 32;
        {   // cooperative staging: 32 keys x 64 dh of K and V
            const int key = tid >> 3, d0 = (tid & 7) * 8;
            *(bf16x8*)&Ks[key * 72 + d0] =
                *(const bf16x8*)(Kb + base + (size_t)(key0 + key) * DM + d0);
            bf16x8 vv = *(const bf16x8*)(Vb + base + (size_t)(key0 + key) * DM + d0);
#pragma unroll
            for (int j = 0; j < 8; ++j) Vt[(d0 + j) * 40 + key] = vv[j];
        }
        __syncthreads();

        if (key0 <= q0 + 15) {                   // causal skip, barrier-uniform
            v8f sc0 = {}, sc1 = {};
#pragma unroll
            for (int ch = 0; ch < 2; ++ch) {
                Frag16 kf0, kf1;                 // B frag: col = key, contig dh
                const __bf16* kp0 = &Ks[r * 72 + ch * 32 + hi * 16];
                const __bf16* kp1 = &Ks[(16 + r) * 72 + ch * 32 + hi * 16];
                kf0.h[0] = *(const bf16x8*)kp0; kf0.h[1] = *(const bf16x8*)(kp0 + 8);
                kf1.h[0] = *(const bf16x8*)kp1; kf1.h[1] = *(const bf16x8*)(kp1 + 8);
                sc0 = __builtin_amdgcn_wmma_f32_16x16x32_bf16(
                    false, qf[ch].v, false, kf0.v, (short)0, sc0, false, false);
                sc1 = __builtin_amdgcn_wmma_f32_16x16x32_bf16(
                    false, qf[ch].v, false, kf1.v, (short)0, sc1, false, false);
            }
            float s0a[8], s1a[8], tm[8];
#pragma unroll
            for (int v = 0; v < 8; ++v) {        // scale + causal mask + rowmax
                const int q = q0 + hi * 8 + v;
                float s0 = sc0[v] * 0.125f;      // 1/sqrt(64)
                float s1 = sc1[v] * 0.125f;
                if (key0 + r > q)      s0 = -1e30f;
                if (key0 + 16 + r > q) s1 = -1e30f;
                float mx = fmaxf(s0, s1);        // butterfly within 16-lane half
                mx = fmaxf(mx, __shfl_xor(mx, 1, 32));
                mx = fmaxf(mx, __shfl_xor(mx, 2, 32));
                mx = fmaxf(mx, __shfl_xor(mx, 4, 32));
                mx = fmaxf(mx, __shfl_xor(mx, 8, 32));
                s0a[v] = s0; s1a[v] = s1; tm[v] = mx;
            }
#pragma unroll
            for (int v = 0; v < 8; ++v) {        // online softmax update
                const float mnew  = fmaxf(mrow[v], tm[v]);
                const float alpha = __expf(mrow[v] - mnew);
                const float p0 = __expf(s0a[v] - mnew);
                const float p1 = __expf(s1a[v] - mnew);
                float rs = p0 + p1;
                rs += __shfl_xor(rs, 1, 32);
                rs += __shfl_xor(rs, 2, 32);
                rs += __shfl_xor(rs, 4, 32);
                rs += __shfl_xor(rs, 8, 32);
                lrow[v] = lrow[v] * alpha + rs;
                mrow[v] = mnew;
#pragma unroll
                for (int j = 0; j < 4; ++j) o[j][v] = o[j][v] * alpha;
                Ps[(w * 16 + hi * 8 + v) * 40 + r]      = f2bf(p0);
                Ps[(w * 16 + hi * 8 + v) * 40 + 16 + r] = f2bf(p1);
            }
            // P: C-layout -> A-layout through per-wave LDS buffer
            Frag16 pf;
            pf.h[0] = *(const bf16x8*)&Ps[(w * 16 + r) * 40 + hi * 16];
            pf.h[1] = *(const bf16x8*)&Ps[(w * 16 + r) * 40 + hi * 16 + 8];
#pragma unroll
            for (int j = 0; j < 4; ++j) {        // PV: B frag col = dh, contig keys
                Frag16 vfr;
                const __bf16* vp = &Vt[(j * 16 + r) * 40 + hi * 16];
                vfr.h[0] = *(const bf16x8*)vp;
                vfr.h[1] = *(const bf16x8*)(vp + 8);
                o[j] = __builtin_amdgcn_wmma_f32_16x16x32_bf16(
                    false, pf.v, false, vfr.v, (short)0, o[j], false, false);
            }
        }
        __syncthreads();
    }

#pragma unroll
    for (int j = 0; j < 4; ++j)
#pragma unroll
        for (int v = 0; v < 8; ++v) {
            const int q = q0 + hi * 8 + v;
            AO[base + (size_t)q * DM + j * 16 + r] = f2bf(o[j][v] / lrow[v]);
        }
}

// ---------------------------------------------------------------------------
// Host-side orchestration
// ---------------------------------------------------------------------------
extern "C" void kernel_launch(void* const* d_in, const int* in_sizes, int n_in,
                              void* d_out, int out_size, void* d_ws, size_t ws_size,
                              hipStream_t stream) {
    (void)in_sizes; (void)n_in; (void)out_size; (void)ws_size;
    const float* x    = (const float*)d_in[0];
    const float* fqkw = (const float*)d_in[1];
    const float* fvw  = (const float*)d_in[2];
    const float* wq   = (const float*)d_in[3];
    const float* wk   = (const float*)d_in[4];
    const float* wv   = (const float*)d_in[5];
    const float* fn   = (const float*)d_in[6];
    const float* rn   = (const float*)d_in[7];
    const float* wo   = (const float*)d_in[8];
    float* out = (float*)d_out;

    // Workspace layout (bytes), with lifetime-based aliasing (~194 MB total):
    char* ws = (char*)d_ws;
    __bf16* Xb   = (__bf16*)(ws);                  // 16 MB  x as bf16
    __bf16* FbT  = (__bf16*)(ws + 16777216);       //  8 MB  f_neurons N x K
    __bf16* RQKT = (__bf16*)(ws + 25165824);       //  4 MB  rqk bank  N x K
    __bf16* RVT  = (__bf16*)(ws + 29360128);       //  4 MB  rv  bank  N x K
    __bf16* WOb  = (__bf16*)(ws + 33554432);       //  2 MB  W_O (already N x K)
    __bf16* Y    = (__bf16*)(ws + 35651584);       // 64 MB  X@F (dead after combine)
    __bf16* Qb   = (__bf16*)(ws + 35651584);       // 16 MB  (aliases Y)
    __bf16* Kb   = (__bf16*)(ws + 52428800);       // 16 MB  (aliases Y)
    __bf16* Vb   = (__bf16*)(ws + 69206016);       // 16 MB  (aliases Y)
    __bf16* AQ   = (__bf16*)(ws + 102760448);      // 32 MB
    __bf16* AK   = (__bf16*)(ws + 136314880);      // 32 MB
    __bf16* AV   = (__bf16*)(ws + 169869312);      // 32 MB
    __bf16* AO   = (__bf16*)(ws + 102760448);      // 16 MB (aliases dead AQ)

    // 1) precision/layout conversion
    cast_f32_to_bf16<<<8388608 / 256, 256, 0, stream>>>(x, Xb, 8388608);
    pack_f_neurons  <<<4194304 / 256, 256, 0, stream>>>(fn, FbT);
    pack_r_neurons  <<<2097152 / 256, 256, 0, stream>>>(rn, RQKT, 0);
    pack_r_neurons  <<<2097152 / 256, 256, 0, stream>>>(rn, RVT, 32);
    cast_f32_to_bf16<<<1048576 / 256, 256, 0, stream>>>(wo, WOb, 1048576);

    // 2) Y = X @ F   (8192 x 4096, K=1024)
    gemm_bf16_kernel<true><<<dim3(32, 64), 256, 0, stream>>>(
        Xb, FbT, nullptr, Y, 8192, 4096, 1024);

    // 3) per-token reduce + build rank-1 token matrices
    combine_kernel<<<8192, 256, 0, stream>>>(Y, fqkw, fvw, wq, wk, wv, AQ, AK, AV);

    // 4) Q/K/V projections (8192 x 1024, K=2048)
    gemm_bf16_kernel<true><<<dim3(8, 64), 256, 0, stream>>>(
        AQ, RQKT, nullptr, Qb, 8192, 1024, 2048);
    gemm_bf16_kernel<true><<<dim3(8, 64), 256, 0, stream>>>(
        AK, RQKT, nullptr, Kb, 8192, 1024, 2048);
    gemm_bf16_kernel<true><<<dim3(8, 64), 256, 0, stream>>>(
        AV, RVT, nullptr, Vb, 8192, 1024, 2048);

    // 5) causal flash attention, 16 heads of 64
    attn_kernel<<<dim3(16, 16, 4), 256, 0, stream>>>(Qb, Kb, Vb, AO);

    // 6) output projection: out = AO @ W_O^T  (f32 result)
    gemm_bf16_kernel<false><<<dim3(8, 64), 256, 0, stream>>>(
        AO, WOb, out, nullptr, 8192, 1024, 1024);
}
